// RotarySelfAttention_51410758533608
// MI455X (gfx1250) — matrix operand order repaired and checked by reading, not compiled, not run on hardware
//
#include <hip/hip_runtime.h>
#include <hip/hip_bf16.h>
#include <stdint.h>

// ---------------- CDNA5 WMMA types & helpers ----------------
typedef __attribute__((ext_vector_type(16))) __bf16 v16bf;
typedef __attribute__((ext_vector_type(8)))  float  v8f;

union FragU {
    uint32_t u[8];
    v16bf    v;
};

// A-matrix 16x32 bf16 layout (ISA 7.12.2): lane m=l&15, hi=l>>4.
// VGPR j (pair of bf16): j<4 -> K = 8*hi + 2j ; j>=4 -> K = 16 + 8*hi + 2(j-4)
__device__ __forceinline__ int a_u32_idx(int j, int hi) {
    return (j < 4) ? (4 * hi + j) : (4 + 4 * hi + j); // u32 index within row (K/2)
}
// B-matrix 32x16 bf16 layout: lane col n=l&15; VGPR j: j<4 -> K = 8*hi + 2j ; j>=4 -> K = 16 + 8*hi + 2(j-4)
__device__ __forceinline__ int b_u32_idx(int j, int hi) {
    return (j < 4) ? (4 * hi + j) : (4 + 4 * hi + j);
}

__device__ __forceinline__ v16bf load_frag_a(const __bf16* rowk0, int hi) {
    const uint32_t* p = (const uint32_t*)rowk0;
    FragU f;
#pragma unroll
    for (int j = 0; j < 8; ++j) f.u[j] = p[a_u32_idx(j, hi)];
    return f.v;
}

__device__ __forceinline__ v16bf load_frag_b(const __bf16* colk0, int hi) {
    const uint32_t* p = (const uint32_t*)colk0;
    FragU f;
#pragma unroll
    for (int j = 0; j < 8; ++j) f.u[j] = p[b_u32_idx(j, hi)];
    return f.v;
}

__device__ __forceinline__ v8f wmma_bf16(v16bf a, v16bf b, v8f c) {
    return __builtin_amdgcn_wmma_f32_16x16x32_bf16(false, a, false, b, (short)0, c,
                                                   false, false);
}

// ---------------- Problem constants ----------------
#define BB 2
#define NN 2048
#define DD 1024
#define HH 16
#define DH 64
#define SCALE 0.125f
#define NEGBIG (-3.0e38f)

// ---------------- Kernel 1: LayerNorm -> bf16 ----------------
__global__ __launch_bounds__(256) void ln_kernel(const float* __restrict__ x,
                                                 const float* __restrict__ gamma,
                                                 const float* __restrict__ beta,
                                                 __bf16* __restrict__ xn) {
    __shared__ float rs[8], rs2[8];
    const int row = blockIdx.x;
    const float* xr = x + (size_t)row * DD;
    float s = 0.f, s2 = 0.f;
    for (int i = threadIdx.x; i < DD; i += 256) {
        float v = xr[i];
        s += v;
        s2 += v * v;
    }
#pragma unroll
    for (int m = 16; m >= 1; m >>= 1) {
        s  += __shfl_xor(s, m, 32);
        s2 += __shfl_xor(s2, m, 32);
    }
    const int wv = threadIdx.x >> 5, ln = threadIdx.x & 31;
    if (ln == 0) { rs[wv] = s; rs2[wv] = s2; }
    __syncthreads();
    if (threadIdx.x == 0) {
        float a = 0.f, b = 0.f;
#pragma unroll
        for (int i = 0; i < 8; ++i) { a += rs[i]; b += rs2[i]; }
        rs[0] = a; rs2[0] = b;
    }
    __syncthreads();
    const float mu  = rs[0] * (1.0f / DD);
    const float var = rs2[0] * (1.0f / DD) - mu * mu;
    const float inv = rsqrtf(var + 1e-5f);
    for (int i = threadIdx.x; i < DD; i += 256) {
        float v = (xr[i] - mu) * inv * gamma[i] + beta[i];
        xn[(size_t)row * DD + i] = (__bf16)v;
    }
}

// ---------------- Kernel 2: weight transpose + bf16 convert ----------------
__global__ __launch_bounds__(256) void wconv_kernel(const float* __restrict__ wqkv,
                                                    const float* __restrict__ wout,
                                                    __bf16* __restrict__ wqkvT,
                                                    __bf16* __restrict__ woutT) {
    const int idx = blockIdx.x * 256 + threadIdx.x;
    const int NQ = 3072 * 1024;
    if (idx < NQ) {
        int n = idx >> 10, k = idx & 1023;
        wqkvT[idx] = (__bf16)wqkv[(size_t)k * 3072 + n];
    } else {
        int i2 = idx - NQ;
        if (i2 < 1024 * 1024) {
            int j = i2 >> 10, k = i2 & 1023;
            woutT[i2] = (__bf16)wout[(size_t)k * 1024 + j];
        }
    }
}

// ---------------- Kernel 3/6: bf16 WMMA GEMM, 64x64 tile per wave ----------------
// C[M,Nc] = A[M,K] (bf16 row-major) * Bt[Nc,K]^T (bf16, weights stored K-contig).
// OutT selects fp32 (final projection, + bias) or bf16 (qkv intermediate).
template <typename OutT, bool HasBias>
__global__ __launch_bounds__(128) void gemm_bf16_kernel(const __bf16* __restrict__ A,
                                                        const __bf16* __restrict__ Bt,
                                                        OutT* __restrict__ C,
                                                        const float* __restrict__ bias,
                                                        int M, int K, int Nc) {
    const int lane = threadIdx.x & 31;
    const int wave = threadIdx.x >> 5;
    const int m16 = lane & 15, hi = lane >> 4;
    const int m0 = blockIdx.x * 64;
    const int n0 = (blockIdx.y * 4 + wave) * 64;

    v8f acc[4][4];
#pragma unroll
    for (int i = 0; i < 4; ++i)
#pragma unroll
        for (int j = 0; j < 4; ++j) acc[i][j] = (v8f)(0.f);

    for (int k0 = 0; k0 < K; k0 += 32) {
        v16bf af[4], bf[4];
#pragma unroll
        for (int i = 0; i < 4; ++i)
            af[i] = load_frag_a(A + (size_t)(m0 + i * 16 + m16) * K + k0, hi);
#pragma unroll
        for (int j = 0; j < 4; ++j)
            bf[j] = load_frag_b(Bt + (size_t)(n0 + j * 16 + m16) * K + k0, hi);
#pragma unroll
        for (int i = 0; i < 4; ++i)
#pragma unroll
            for (int j = 0; j < 4; ++j) acc[i][j] = wmma_bf16(af[i], bf[j], acc[i][j]);
    }

#pragma unroll
    for (int i = 0; i < 4; ++i)
#pragma unroll
        for (int j = 0; j < 4; ++j) {
            const int col = n0 + j * 16 + m16;
            const float badd = HasBias ? bias[col] : 0.f;
#pragma unroll
            for (int r = 0; r < 8; ++r) {
                const int row = m0 + i * 16 + r + 8 * hi;
                C[(size_t)row * Nc + col] = (OutT)(acc[i][j][r] + badd);
            }
        }
}

// ---------------- Kernel 4: RoPE + head layout (Q,K roped; V transposed) ----------------
__global__ __launch_bounds__(256) void rope_kernel(const __bf16* __restrict__ qkv,
                                                   const float* __restrict__ rot,
                                                   __bf16* __restrict__ Qb,
                                                   __bf16* __restrict__ Kb,
                                                   __bf16* __restrict__ Vt) {
    const int idx = blockIdx.x * 256 + threadIdx.x; // over B*N*H*DH
    const int d = idx & 63;
    const int h = (idx >> 6) & 15;
    const int n = (idx >> 10) & 2047;
    const int b = idx >> 21;
    const size_t row = (size_t)(b * NN + n) * 3072;
    const float f = rot[(size_t)(b * NN + n) * DH + d];
    const float cs = __cosf(f), sn = __sinf(f);
    const int qc = h * DH + d;
    const float qv = (float)qkv[row + qc];
    const float kv = (float)qkv[row + 1024 + qc];
    const float vv = (float)qkv[row + 2048 + qc];
    // rotate_half: r[d] = (d<32) ? -x[2d+1] : x[2(d-32)]
    const int pd = (d < 32) ? (2 * d + 1) : (2 * (d - 32));
    const float sgn = (d < 32) ? -1.f : 1.f;
    const float qp = (float)qkv[row + h * DH + pd];
    const float kp = (float)qkv[row + 1024 + h * DH + pd];
    const size_t bh = (size_t)(b * HH + h);
    Qb[(bh * NN + n) * DH + d] = (__bf16)(qv * cs + sgn * qp * sn);
    Kb[(bh * NN + n) * DH + d] = (__bf16)(kv * cs + sgn * kp * sn);
    Vt[(bh * DH + d) * NN + n] = (__bf16)vv;
}

// ---------------- Kernel 5: flash attention, 64 queries/block, 4 waves ----------------
__global__ __launch_bounds__(128) void attn_kernel(const __bf16* __restrict__ Qb,
                                                   const __bf16* __restrict__ Kb,
                                                   const __bf16* __restrict__ Vt,
                                                   const unsigned char* __restrict__ mask,
                                                   __bf16* __restrict__ attnout) {
    __shared__ __bf16 plds[4][16 * 72]; // per-wave 16x64 P tile, stride 72 (pad)
    const int lane = threadIdx.x & 31;
    const int wave = threadIdx.x >> 5;
    const int m16 = lane & 15, hi = lane >> 4;
    const int b = blockIdx.z, h = blockIdx.y;
    const int q0 = blockIdx.x * 64 + wave * 16;
    const size_t bh = (size_t)(b * HH + h);

    // Q fragments: 16 rows x 64 (two K-steps of 32)
    const __bf16* Qrow = Qb + (bh * NN + q0 + m16) * DH;
    v16bf qf[2];
    qf[0] = load_frag_a(Qrow, hi);
    qf[1] = load_frag_a(Qrow + 32, hi);

    float mrow[8], lrow[8];
    v8f o[4];
#pragma unroll
    for (int r = 0; r < 8; ++r) { mrow[r] = NEGBIG; lrow[r] = 0.f; }
#pragma unroll
    for (int t = 0; t < 4; ++t) o[t] = (v8f)(0.f);

    for (int kb = 0; kb < NN; kb += 64) {
        // S = Q * K^T for a 16x64 tile
        v8f s[4];
#pragma unroll
        for (int t = 0; t < 4; ++t) s[t] = (v8f)(0.f);
#pragma unroll
        for (int t = 0; t < 4; ++t) {
            const __bf16* Krow = Kb + (bh * NN + kb + t * 16 + m16) * DH;
#pragma unroll
            for (int kk = 0; kk < 2; ++kk) {
                v16bf kfrag = load_frag_b(Krow + kk * 32, hi);
                s[t] = wmma_bf16(qf[kk], kfrag, s[t]);
            }
        }
        // per-lane column mask additive term
        float madd[4];
#pragma unroll
        for (int t = 0; t < 4; ++t)
            madd[t] = mask[b * NN + kb + t * 16 + m16] ? 0.f : NEGBIG;

        // scale+mask, row max across half-wave (rows live in lanes hi*16..hi*16+15)
        float mnew[8], alpha[8];
#pragma unroll
        for (int r = 0; r < 8; ++r) {
            float mx = NEGBIG;
#pragma unroll
            for (int t = 0; t < 4; ++t) {
                float sv = s[t][r] * SCALE + madd[t];
                s[t][r] = sv;
                mx = fmaxf(mx, sv);
            }
            mx = fmaxf(mx, __shfl_xor(mx, 1, 32));
            mx = fmaxf(mx, __shfl_xor(mx, 2, 32));
            mx = fmaxf(mx, __shfl_xor(mx, 4, 32));
            mx = fmaxf(mx, __shfl_xor(mx, 8, 32));
            mnew[r] = fmaxf(mrow[r], mx);
            alpha[r] = (mrow[r] <= -1e37f) ? 0.f : __expf(mrow[r] - mnew[r]);
            mrow[r] = mnew[r];
        }

        __syncthreads();
        // P = exp(S - m), row sums; stage bf16 P to LDS (C-layout -> A-layout)
#pragma unroll
        for (int r = 0; r < 8; ++r) {
            const int rowm = r + 8 * hi;
            float ps = 0.f;
#pragma unroll
            for (int t = 0; t < 4; ++t) {
                float p = __expf(s[t][r] - mnew[r]);
                ps += p;
                plds[wave][rowm * 72 + t * 16 + m16] = (__bf16)p;
            }
            ps += __shfl_xor(ps, 1, 32);
            ps += __shfl_xor(ps, 2, 32);
            ps += __shfl_xor(ps, 4, 32);
            ps += __shfl_xor(ps, 8, 32);
            lrow[r] = lrow[r] * alpha[r] + ps;
        }
        __syncthreads();

        // rescale running O
#pragma unroll
        for (int t = 0; t < 4; ++t)
#pragma unroll
            for (int r = 0; r < 8; ++r) o[t][r] *= alpha[r];

        // P fragments from LDS (A-layout reads)
        v16bf pf[2];
        pf[0] = load_frag_a(&plds[wave][m16 * 72], hi);
        pf[1] = load_frag_a(&plds[wave][m16 * 72] + 32, hi);

        // O += P * V  (V B-fragments from transposed layout, K-contiguous)
#pragma unroll
        for (int t = 0; t < 4; ++t) {
            const __bf16* Vrow = Vt + (bh * DH + t * 16 + m16) * NN + kb;
#pragma unroll
            for (int kk = 0; kk < 2; ++kk) {
                v16bf vfrag = load_frag_b(Vrow + kk * 32, hi);
                o[t] = wmma_bf16(pf[kk], vfrag, o[t]);
            }
        }
        __syncthreads();
    }

    // normalize and write bf16 attention output [B,N,H*DH]
#pragma unroll
    for (int t = 0; t < 4; ++t)
#pragma unroll
        for (int r = 0; r < 8; ++r) {
            const int qrow = q0 + r + 8 * hi;
            const int col = h * DH + t * 16 + m16;
            float val = o[t][r] / lrow[r];
            attnout[(size_t)(b * NN + qrow) * (HH * DH) + col] = (__bf16)val;
        }
}

// ---------------- Host launch ----------------
extern "C" void kernel_launch(void* const* d_in, const int* in_sizes, int n_in,
                              void* d_out, int out_size, void* d_ws, size_t ws_size,
                              hipStream_t stream) {
    const float* x     = (const float*)d_in[0];
    const float* rot   = (const float*)d_in[1];
    const unsigned char* mask = (const unsigned char*)d_in[2];
    const float* gamma = (const float*)d_in[3];
    const float* beta  = (const float*)d_in[4];
    const float* wqkv  = (const float*)d_in[5];
    const float* wout  = (const float*)d_in[6];
    const float* bout  = (const float*)d_in[7];
    float* out = (float*)d_out;

    char* ws = (char*)d_ws;
    __bf16* xn      = (__bf16*)(ws);                       //  8 MiB [4096,1024]
    __bf16* wqkvT   = (__bf16*)(ws + (size_t)(8u  << 20)); //  6 MiB [3072,1024]
    __bf16* woutT   = (__bf16*)(ws + (size_t)(14u << 20)); //  2 MiB [1024,1024]
    __bf16* qkv     = (__bf16*)(ws + (size_t)(16u << 20)); // 24 MiB [4096,3072]
    __bf16* Qb      = (__bf16*)(ws + (size_t)(40u << 20)); //  8 MiB [B,H,N,DH]
    __bf16* Kb      = (__bf16*)(ws + (size_t)(48u << 20)); //  8 MiB [B,H,N,DH]
    __bf16* Vt      = (__bf16*)(ws + (size_t)(56u << 20)); //  8 MiB [B,H,DH,N]
    __bf16* attnout = (__bf16*)(ws + (size_t)(64u << 20)); //  8 MiB [4096,1024]

    ln_kernel<<<BB * NN, 256, 0, stream>>>(x, gamma, beta, xn);

    const int wtotal = 3072 * 1024 + 1024 * 1024;
    wconv_kernel<<<(wtotal + 255) / 256, 256, 0, stream>>>(wqkv, wout, wqkvT, woutT);

    // QKV GEMM: [4096,1024] x [1024,3072] -> bf16
    gemm_bf16_kernel<__bf16, false><<<dim3(4096 / 64, 3072 / 256), 128, 0, stream>>>(
        xn, wqkvT, qkv, nullptr, BB * NN, DD, 3 * HH * DH);

    rope_kernel<<<(BB * NN * HH * DH) / 256, 256, 0, stream>>>(qkv, rot, Qb, Kb, Vt);

    attn_kernel<<<dim3(NN / 64, HH, BB), 128, 0, stream>>>(Qb, Kb, Vt, mask, attnout);

    // Output GEMM + bias: [4096,1024] x [1024,1024] -> fp32
    gemm_bf16_kernel<float, true><<<dim3(4096 / 64, 1024 / 256), 128, 0, stream>>>(
        attnout, woutT, out, bout, BB * NN, HH * DH, DD);
}
